// AttentionSharingUnit_74577812128127
// MI455X (gfx1250) — matrix operand order repaired and checked by reading, not compiled, not run on hardware
//
#include <hip/hip_runtime.h>

typedef __attribute__((ext_vector_type(16))) __bf16 v16bf;
typedef __attribute__((ext_vector_type(8)))  float  v8f;

#define HEADS 20
#define CDIM  1280
#define RANK  256
#define DDIM  2048
#define MROWS 8192   /* B*FRAMES*D = 2*2*2048 */
#define DH    64

// GEMM LDS staging geometry (bf16 elements)
#define GPAD   40                      // padded row length for a 32-wide K slab
#define ATILE  (128 * GPAD)            // 5120 u16 = 10240 B
#define STAGE  (2 * ATILE)             // A + B slab = 20480 B
// dynamic LDS total = 2 * STAGE * 2 bytes = 40960 B

// ---------------------------------------------------------------- helpers
__device__ __forceinline__ unsigned short f2bf(float f) {
  unsigned u = __float_as_uint(f);
  u += 0x7FFFu + ((u >> 16) & 1u);           // round-to-nearest-even
  return (unsigned short)(u >> 16);
}
__device__ __forceinline__ float bf2f(unsigned short s) {
  return __uint_as_float(((unsigned)s) << 16);
}
__device__ __forceinline__ v8f wmma_bf16(v16bf a, v16bf b, v8f c) {
  return __builtin_amdgcn_wmma_f32_16x16x32_bf16(false, a, false, b, (short)0, c,
                                                 false, false);
}
// gfx1250 async copy: global -> LDS, 16B per lane, tracked by ASYNCcnt
__device__ __forceinline__ void async_ld16(unsigned lds_byte_off, const void* gaddr) {
  asm volatile("global_load_async_to_lds_b128 %0, %1, off"
               :: "v"(lds_byte_off), "v"(gaddr) : "memory");
}
__device__ __forceinline__ void wait_async_le8() {
  asm volatile("s_wait_asynccnt 0x8" ::: "memory");
}
__device__ __forceinline__ void wait_async_0() {
  asm volatile("s_wait_asynccnt 0x0" ::: "memory");
}

// ---------------------------------------------------------------- f32 -> bf16
__global__ void k_cvt(const float* __restrict__ s, unsigned short* __restrict__ d, int n) {
  int i = blockIdx.x * blockDim.x + threadIdx.x;
  if (i < n) d[i] = f2bf(s[i]);
}

// ---------------------------------------------------------------- elementwise
__global__ void k_add(const float* __restrict__ a, const float* __restrict__ b,
                      float* __restrict__ o, int n) {
  int i = blockIdx.x * blockDim.x + threadIdx.x;
  if (i < n) o[i] = a[i] + b[i];
}
__global__ void k_addsub(const float* __restrict__ a, const float* __restrict__ b,
                         const float* __restrict__ c, float* __restrict__ o, int n) {
  int i = blockIdx.x * blockDim.x + threadIdx.x;
  if (i < n) o[i] = a[i] + b[i] - c[i];
}

// ---------------------------------------------------------------- GEMM (NT)
// C[M,N] = A[M,K] * B[N,K]^T  (bf16 in, f32 WMMA accumulate)
// 128x128 block tile, 4 waves of 64x64, K-step 32, async double-buffered LDS.
// optional: += (accumulate), bias[N], bf16 out, per-frame B (frame=(row/2048)&1)
__global__ __launch_bounds__(128)
void k_gemm_nt(const unsigned short* __restrict__ A,
               const unsigned short* __restrict__ B,
               float* __restrict__ Cf,
               unsigned short* __restrict__ Cb,
               const float* __restrict__ bias,
               int M, int N, int K, int accumulate, int outBf,
               long frameStrideB)
{
  extern __shared__ __align__(16) unsigned short smem[];   // 2 stages of A+B slabs
  const int m0 = blockIdx.y * 128;
  const int n0 = blockIdx.x * 128;
  const unsigned short* Bp =
      B + (frameStrideB ? ((long)((m0 >> 11) & 1)) * frameStrideB : 0);
  const int tid = threadIdx.x;
  const int w = tid >> 5, lane = tid & 31, half = lane >> 4, l15 = lane & 15;
  const int wm = (w >> 1) * 64, wn = (w & 1) * 64;

  const v8f vz = {0.f,0.f,0.f,0.f,0.f,0.f,0.f,0.f};
  v8f acc[4][4];
  #pragma unroll
  for (int mi = 0; mi < 4; ++mi)
    #pragma unroll
    for (int ni = 0; ni < 4; ++ni) acc[mi][ni] = vz;

  // issue one 128x32 A slab + 128x32 B slab into stage p (8 async ops per wave)
  auto issue = [&](int p, int k0) {
    const unsigned ab = (unsigned)(p * STAGE) * 2;           // byte offsets in LDS
    const unsigned bb = ab + (unsigned)ATILE * 2;
    #pragma unroll
    for (int i = 0; i < 4; ++i) {
      int idx = tid + i * 128;                 // 0..511 uint4 chunks per slab
      int row = idx >> 2, c8 = (idx & 3) * 8;  // 4 chunks per 32-u16 row
      async_ld16(ab + (unsigned)(row * GPAD + c8) * 2,
                 A  + (size_t)(m0 + row) * K + k0 + c8);
      async_ld16(bb + (unsigned)(row * GPAD + c8) * 2,
                 Bp + (size_t)(n0 + row) * K + k0 + c8);
    }
  };

  issue(0, 0);
  int p = 0;
  for (int k0 = 0; k0 < K; k0 += 32) {
    const int hasNext = (k0 + 32 < K);
    if (hasNext) issue(p ^ 1, k0 + 32);
    if (hasNext) wait_async_le8(); else wait_async_0();  // stage-p data landed
    __syncthreads();                                     // ... for every wave

    const unsigned short* As = smem + p * STAGE;
    const unsigned short* Bs = As + ATILE;
    union { v16bf v; unsigned u[8]; } fa[4], fb[4];
    #pragma unroll
    for (int mi = 0; mi < 4; ++mi) {          // A frag: 16x32 bf16 (ISA layout)
      const unsigned short* ap = &As[(wm + mi*16 + l15) * GPAD + half * 8];
      #pragma unroll
      for (int v = 0; v < 8; ++v)
        fa[mi].u[v] = *(const unsigned*)(ap + ((v & 4) << 2) + ((v & 3) << 1));
    }
    #pragma unroll
    for (int ni = 0; ni < 4; ++ni) {          // B frag: 32x16, K down halfwords
      const unsigned short* bp = &Bs[(wn + ni*16 + l15) * GPAD + half * 8];
      #pragma unroll
      for (int t = 0; t < 8; ++t)
        fb[ni].u[t] = *(const unsigned*)(bp + ((t & 4) << 2) + ((t & 3) << 1));
    }
    #pragma unroll
    for (int mi = 0; mi < 4; ++mi)
      #pragma unroll
      for (int ni = 0; ni < 4; ++ni)
        acc[mi][ni] = wmma_bf16(fa[mi].v, fb[ni].v, acc[mi][ni]);

    __syncthreads();    // all reads of stage p done before it is refilled
    p ^= 1;
  }

  #pragma unroll
  for (int mi = 0; mi < 4; ++mi)
    #pragma unroll
    for (int ni = 0; ni < 4; ++ni)
      #pragma unroll
      for (int v = 0; v < 8; ++v) {
        int row = m0 + wm + mi*16 + v + 8*half;
        int col = n0 + wn + ni*16 + l15;
        size_t off = (size_t)row * N + col;
        float val = acc[mi][ni][v];
        if (bias) val += bias[col];
        if (outBf) {
          if (accumulate) val += bf2f(Cb[off]);
          Cb[off] = f2bf(val);
        } else {
          if (accumulate) val += Cf[off];
          Cf[off] = val;
        }
      }
}

// ---------------------------------------------------------------- flash attention
// grid = (2048/64, HEADS, 4), block 128. Q,K,V,O bf16 [4*2048, 1280].
__global__ __launch_bounds__(128)
void k_flash_attn(const unsigned short* __restrict__ Q,
                  const unsigned short* __restrict__ Kt,
                  const unsigned short* __restrict__ Vv,
                  unsigned short* __restrict__ O)
{
  __shared__ __align__(16) unsigned short Ks[64][72];     // [kv][dh]
  __shared__ __align__(16) unsigned short Vt[64][72];     // [dh][kv] (transposed)
  __shared__ __align__(16) unsigned short Ps[4][16][72];  // per-wave P tiles

  const int qt0  = blockIdx.x * 64;
  const int head = blockIdx.y;
  const int bf   = blockIdx.z;
  const int tid = threadIdx.x, w = tid >> 5, lane = tid & 31;
  const int half = lane >> 4, l15 = lane & 15;
  const size_t base = ((size_t)bf * DDIM) * CDIM + head * DH;

  // Q fragments for this wave's 16 rows (kept in registers)
  v16bf qf[2];
  {
    const unsigned short* qp = Q + base + (size_t)(qt0 + w*16 + l15) * CDIM;
    #pragma unroll
    for (int kk2 = 0; kk2 < 2; ++kk2) {
      union { v16bf v; unsigned u[8]; } fa;
      #pragma unroll
      for (int v = 0; v < 8; ++v)
        fa.u[v] = *(const unsigned*)(qp + kk2*32 + ((v & 4) << 2) + ((v & 3) << 1)
                                     + half * 8);
      qf[kk2] = fa.v;
    }
  }

  const v8f vz = {0.f,0.f,0.f,0.f,0.f,0.f,0.f,0.f};
  v8f o0[4]; o0[0]=vz; o0[1]=vz; o0[2]=vz; o0[3]=vz;
  float rowmax[8], rowsum[8];
  #pragma unroll
  for (int v = 0; v < 8; ++v) { rowmax[v] = -1e30f; rowsum[v] = 0.f; }

  for (int k0 = 0; k0 < DDIM; k0 += 64) {
    #pragma unroll
    for (int r = 0; r < 4; ++r) {                 // stage K and V^T tiles
      int idx = tid + r * 128;
      int row = idx >> 3, c8 = (idx & 7) * 8;
      *(uint4*)&Ks[row][c8] =
          *(const uint4*)(Kt + base + (size_t)(k0 + row) * CDIM + c8);
      union { uint4 u4; unsigned short s[8]; } uv;
      uv.u4 = *(const uint4*)(Vv + base + (size_t)(k0 + row) * CDIM + c8);
      #pragma unroll
      for (int j = 0; j < 8; ++j) Vt[c8 + j][row] = uv.s[j];
    }
    __syncthreads();

    // S = Q K^T (16 x 64 per wave)
    v8f s[4]; s[0]=vz; s[1]=vz; s[2]=vz; s[3]=vz;
    #pragma unroll
    for (int kk2 = 0; kk2 < 2; ++kk2)
      #pragma unroll
      for (int nt = 0; nt < 4; ++nt) {
        union { v16bf v; unsigned u[8]; } fb;
        const unsigned short* kp = &Ks[nt*16 + l15][kk2*32 + half*8];
        #pragma unroll
        for (int t = 0; t < 8; ++t) fb.u[t] = *(const unsigned*)(kp + ((t & 4) << 2) + ((t & 3) << 1));
        s[nt] = wmma_bf16(qf[kk2], fb.v, s[nt]);
      }
    #pragma unroll
    for (int nt = 0; nt < 4; ++nt)
      #pragma unroll
      for (int v = 0; v < 8; ++v)
        s[nt][v] *= 0.125f;                       // dh^-0.5

    // online softmax (row reductions over 16-lane groups)
    #pragma unroll
    for (int v = 0; v < 8; ++v) {
      float pm = fmaxf(fmaxf(s[0][v], s[1][v]), fmaxf(s[2][v], s[3][v]));
      pm = fmaxf(pm, __shfl_xor(pm, 1, 16));
      pm = fmaxf(pm, __shfl_xor(pm, 2, 16));
      pm = fmaxf(pm, __shfl_xor(pm, 4, 16));
      pm = fmaxf(pm, __shfl_xor(pm, 8, 16));
      float mn = fmaxf(rowmax[v], pm);
      float alpha = __expf(rowmax[v] - mn);
      rowmax[v] = mn;
      float sum = 0.f;
      #pragma unroll
      for (int nt = 0; nt < 4; ++nt) {
        float pv = __expf(s[nt][v] - mn);
        sum += pv;
        Ps[w][v + 8*half][nt*16 + l15] = f2bf(pv);
      }
      sum += __shfl_xor(sum, 1, 16);
      sum += __shfl_xor(sum, 2, 16);
      sum += __shfl_xor(sum, 4, 16);
      sum += __shfl_xor(sum, 8, 16);
      rowsum[v] = rowsum[v] * alpha + sum;
      #pragma unroll
      for (int dt = 0; dt < 4; ++dt) o0[dt][v] *= alpha;
    }

    // O += P V
    #pragma unroll
    for (int kk2 = 0; kk2 < 2; ++kk2) {
      union { v16bf v; unsigned u[8]; } fa;
      const unsigned short* pp = &Ps[w][l15][0];
      #pragma unroll
      for (int v = 0; v < 8; ++v)
        fa.u[v] = *(const unsigned*)(pp + kk2*32 + ((v & 4) << 2) + ((v & 3) << 1)
                                     + half * 8);
      #pragma unroll
      for (int dt = 0; dt < 4; ++dt) {
        union { v16bf v; unsigned u[8]; } fb;
        const unsigned short* vp = &Vt[dt*16 + l15][kk2*32 + half*8];
        #pragma unroll
        for (int t = 0; t < 8; ++t) fb.u[t] = *(const unsigned*)(vp + ((t & 4) << 2) + ((t & 3) << 1));
        o0[dt] = wmma_bf16(fa.v, fb.v, o0[dt]);
      }
    }
    __syncthreads();
  }

  #pragma unroll
  for (int v = 0; v < 8; ++v) {
    float inv = 1.0f / rowsum[v];
    size_t ro = base + (size_t)(qt0 + w*16 + v + 8*half) * CDIM;
    #pragma unroll
    for (int dt = 0; dt < 4; ++dt)
      O[ro + dt*16 + l15] = f2bf(o0[dt][v] * inv);
  }
}

// ---------------------------------------------------------------- LayerNorm (row/block)
__global__ __launch_bounds__(256)
void k_layernorm(const float* __restrict__ x, const float* __restrict__ g,
                 const float* __restrict__ bb, unsigned short* __restrict__ out)
{
  const int row = blockIdx.x;
  const float* xr = x + (size_t)row * CDIM;
  float s = 0.f, s2 = 0.f;
  for (int c = threadIdx.x; c < CDIM; c += 256) { float v = xr[c]; s += v; s2 += v*v; }
  #pragma unroll
  for (int o = 16; o > 0; o >>= 1) { s += __shfl_xor(s, o, 32); s2 += __shfl_xor(s2, o, 32); }
  __shared__ float ss[8], ss2[8];
  if ((threadIdx.x & 31) == 0) { ss[threadIdx.x >> 5] = s; ss2[threadIdx.x >> 5] = s2; }
  __syncthreads();
  float ts = 0.f, ts2 = 0.f;
  #pragma unroll
  for (int i = 0; i < 8; ++i) { ts += ss[i]; ts2 += ss2[i]; }
  const float mu  = ts * (1.0f / CDIM);
  const float var = ts2 * (1.0f / CDIM) - mu * mu;
  const float inv = rsqrtf(var + 1e-6f);
  for (int c = threadIdx.x; c < CDIM; c += 256)
    out[(size_t)row * CDIM + c] = f2bf((xr[c] - mu) * inv * g[c] + bb[c]);
}

// ---------------------------------------------------------------- temporal attention (f=2)
__global__ void k_tattn_logits(const unsigned short* __restrict__ qt,
                               const unsigned short* __restrict__ kt,
                               float* __restrict__ probs)
{
  int idx = blockIdx.x * blockDim.x + threadIdx.x;    // (b*d)*heads
  if (idx >= 4096 * HEADS) return;
  int head = idx % HEADS, bd = idx / HEADS;
  int b = bd >> 11, dd = bd & 2047;
  size_t r0 = ((size_t)(b*2 + 0) * DDIM + dd) * CDIM + head * DH;
  size_t r1 = ((size_t)(b*2 + 1) * DDIM + dd) * CDIM + head * DH;
  float l00 = 0.f, l01 = 0.f, l10 = 0.f, l11 = 0.f;
  for (int k = 0; k < DH; ++k) {
    float q0 = bf2f(qt[r0 + k]), q1 = bf2f(qt[r1 + k]);
    float c0 = bf2f(kt[r0 + k]), c1 = bf2f(kt[r1 + k]);
    l00 += q0*c0; l01 += q0*c1; l10 += q1*c0; l11 += q1*c1;
  }
  const float sc = 0.125f;
  l00 *= sc; l01 *= sc; l10 *= sc; l11 *= sc;
  float m0 = fmaxf(l00, l01), e0 = __expf(l00 - m0), e1 = __expf(l01 - m0);
  float i0 = 1.f / (e0 + e1);
  float m1 = fmaxf(l10, l11), e2 = __expf(l10 - m1), e3 = __expf(l11 - m1);
  float i1 = 1.f / (e2 + e3);
  float* p = probs + (size_t)idx * 4;
  p[0] = e0*i0; p[1] = e1*i0; p[2] = e2*i1; p[3] = e3*i1;
}

__global__ void k_tattn_out(const unsigned short* __restrict__ vt,
                            const float* __restrict__ probs,
                            unsigned short* __restrict__ out)
{
  size_t idx = (size_t)blockIdx.x * blockDim.x + threadIdx.x;
  if (idx >= (size_t)MROWS * CDIM) return;
  int col = (int)(idx % CDIM);
  size_t row = idx / CDIM;
  int bfi = (int)(row >> 11), b = bfi >> 1, fr = bfi & 1, dd = (int)(row & 2047);
  int head = col >> 6, bd = (b << 11) + dd;
  const float* p = probs + ((size_t)(bd * HEADS + head)) * 4 + fr * 2;
  float v0 = bf2f(vt[((size_t)(b*2 + 0) * DDIM + dd) * CDIM + col]);
  float v1 = bf2f(vt[((size_t)(b*2 + 1) * DDIM + dd) * CDIM + col]);
  out[idx] = f2bf(p[0]*v0 + p[1]*v1);
}

// ---------------------------------------------------------------- host orchestration
extern "C" void kernel_launch(void* const* d_in, const int* in_sizes, int n_in,
                              void* d_out, int out_size, void* d_ws, size_t ws_size,
                              hipStream_t stream)
{
  (void)in_sizes; (void)n_in; (void)out_size; (void)ws_size;
  const float* h     = (const float*)d_in[0];
  const float* bo    = (const float*)d_in[5];
  const float* gamma = (const float*)d_in[14];
  const float* beta  = (const float*)d_in[15];
  const float* bi    = (const float*)d_in[17];
  const float* btq   = (const float*)d_in[19];
  const float* btk   = (const float*)d_in[21];
  const float* btv   = (const float*)d_in[23];
  const float* bto   = (const float*)d_in[25];

  const size_t MC = (size_t)MROWS * CDIM;       // activation elements
  const size_t WC = (size_t)CDIM * CDIM;        // weight elements
  const size_t LC = (size_t)2 * RANK * CDIM;    // lora tensor elements (2 frames)
  const size_t RC = (size_t)MROWS * RANK;

  char* wp = (char*)d_ws;
  auto take = [&](size_t bytes) -> char* {
    char* r = wp; wp += (bytes + 255) & ~(size_t)255; return r;
  };
  unsigned short* hbf  = (unsigned short*)take(MC * 2);
  unsigned short* wbf  = (unsigned short*)take(9 * WC * 2);  // q,k,v,o,i,tq,tk,tv,to
  unsigned short* lbf  = (unsigned short*)take(8 * LC * 2);  // Dq,Uq,Dk,Uk,Dv,Uv,Do,Uo
  unsigned short* rbf  = (unsigned short*)take(RC * 2);
  unsigned short* qbf  = (unsigned short*)take(MC * 2);
  unsigned short* kbf  = (unsigned short*)take(MC * 2);
  unsigned short* vbf  = (unsigned short*)take(MC * 2);
  unsigned short* obf  = (unsigned short*)take(MC * 2);
  unsigned short* xnbf = (unsigned short*)take(MC * 2);
  unsigned short* xibf = (unsigned short*)take(MC * 2);
  unsigned short* xobf = (unsigned short*)take(MC * 2);
  float* bufA  = (float*)take(MC * 4);
  float* bufMH = (float*)take(MC * 4);
  float* probs = (float*)take((size_t)4096 * HEADS * 4 * 4);

  auto cvt = [&](const float* src, unsigned short* dst, size_t n) {
    k_cvt<<<dim3((unsigned)((n + 255) / 256)), dim3(256), 0, stream>>>(src, dst, (int)n);
  };
  const unsigned gemmLds = 2 * STAGE * 2;   // 40960 B, double-buffered slabs
  auto gemm = [&](const unsigned short* A, const unsigned short* B, float* Cf,
                  unsigned short* Cb, const float* bias, int M, int N, int K,
                  int acc, int outBf, long fs) {
    k_gemm_nt<<<dim3(N / 128, M / 128), dim3(128), gemmLds, stream>>>(
        A, B, Cf, Cb, bias, M, N, K, acc, outBf, fs);
  };

  // ---- 0: convert everything to bf16
  cvt(h, hbf, MC);
  const int widx[9] = {1, 2, 3, 4, 16, 18, 20, 22, 24};
  for (int i = 0; i < 9; ++i) cvt((const float*)d_in[widx[i]], wbf + i * WC, WC);
  const int lidx[8] = {6, 7, 8, 9, 10, 11, 12, 13};
  for (int i = 0; i < 8; ++i) cvt((const float*)d_in[lidx[i]], lbf + i * LC, LC);

  const long FS_D = (long)RANK * CDIM;   // per-frame stride in D [2,R,C]
  const long FS_U = (long)CDIM * RANK;   // per-frame stride in U [2,C,R]

  // ---- 1: spatial q/k/v with LoRA
  unsigned short* qkv[3] = {qbf, kbf, vbf};
  for (int pI = 0; pI < 3; ++pI) {
    gemm(hbf, lbf + (2*pI + 0) * LC, nullptr, rbf, nullptr,
         MROWS, RANK, CDIM, 0, 1, FS_D);                         // r = x D^T
    gemm(hbf, wbf + pI * WC, bufA, nullptr, nullptr,
         MROWS, CDIM, CDIM, 0, 0, 0);                            // y = x W^T
    gemm(rbf, lbf + (2*pI + 1) * LC, bufA, nullptr, nullptr,
         MROWS, CDIM, RANK, 1, 0, FS_U);                         // y += r U^T
    cvt(bufA, qkv[pI], MC);
  }

  // ---- 2: spatial flash attention (per bf slice, per head)
  k_flash_attn<<<dim3(DDIM / 64, HEADS, 4), dim3(128), 0, stream>>>(qbf, kbf, vbf, obf);

  // ---- 3: output projection with LoRA + bias + residual
  gemm(obf, lbf + 6 * LC, nullptr, rbf, nullptr, MROWS, RANK, CDIM, 0, 1, FS_D);
  gemm(obf, wbf + 3 * WC, bufA, nullptr, nullptr, MROWS, CDIM, CDIM, 0, 0, 0);
  gemm(rbf, lbf + 7 * LC, bufA, nullptr, bo, MROWS, CDIM, RANK, 1, 0, FS_U);
  k_add<<<dim3((unsigned)((MC + 255) / 256)), dim3(256), 0, stream>>>(
      h, bufA, bufMH, (int)MC);                                  // mh = h + o

  // ---- 4: LayerNorm -> temporal_i
  k_layernorm<<<dim3(MROWS), dim3(256), 0, stream>>>(bufMH, gamma, beta, xnbf);
  gemm(xnbf, wbf + 4 * WC, bufA, nullptr, bi, MROWS, CDIM, CDIM, 0, 0, 0);
  cvt(bufA, xibf, MC);

  // ---- 5: temporal q/k/v (bf16 out, bias in epilogue); reuse q/k/v buffers
  gemm(xibf, wbf + 5 * WC, nullptr, qbf, btq, MROWS, CDIM, CDIM, 0, 1, 0);
  gemm(xibf, wbf + 6 * WC, nullptr, kbf, btk, MROWS, CDIM, CDIM, 0, 1, 0);
  gemm(xibf, wbf + 7 * WC, nullptr, vbf, btv, MROWS, CDIM, CDIM, 0, 1, 0);

  // ---- 6: cross-frame attention (f=2 -> plain VALU)
  k_tattn_logits<<<dim3((4096 * HEADS + 255) / 256), dim3(256), 0, stream>>>(
      qbf, kbf, probs);
  k_tattn_out<<<dim3((unsigned)((MC + 255) / 256)), dim3(256), 0, stream>>>(
      vbf, probs, xobf);

  // ---- 7: temporal_o + final combine: out = mh + xo - h
  gemm(xobf, wbf + 8 * WC, bufA, nullptr, bto, MROWS, CDIM, CDIM, 0, 0, 0);
  k_addsub<<<dim3((unsigned)((MC + 255) / 256)), dim3(256), 0, stream>>>(
      bufMH, bufA, h, (float*)d_out, (int)MC);
}
